// IndRNN_76390288327595
// MI455X (gfx1250) — matrix operand order repaired and checked by reading, not compiled, not run on hardware
//
#include <hip/hip_runtime.h>
#include <hip/hip_bf16.h>

// ---------------------------------------------------------------------------
// IndRNN (depth-2) on MI455X / gfx1250
//   Phase 0: convert X -> bf16, W -> bf16 transposed  (one-shot, L2-resident)
//   Phase 1: P[d] = X @ W[d]  (bf16 WMMA, f32 acc, async global->LDS staging,
//            double-buffered on ASYNCcnt, 64x64 wave tiles -> 512 B LDS/wmma)
//   Phase 2: h = tanh(h*u0 + P0 + b0); h = tanh(h*u1 + P1 + b1) scan over T
// ---------------------------------------------------------------------------

typedef __attribute__((ext_vector_type(16))) __bf16 v16bf;
typedef __attribute__((ext_vector_type(8)))  __bf16 v8bf;
typedef __attribute__((ext_vector_type(4)))  __bf16 v4bf;
typedef __attribute__((ext_vector_type(8)))  float  v8f;

#define B_     64
#define T_     512
#define D_     512
#define R_     1024
#define DEPTH_ 2
#define M_     (B_ * T_)      // 32768 GEMM rows

#define BM  128               // block tile M
#define BN  256               // block tile N
#define BK  32                // K step (one bf16 WMMA K)
#define KT  (D_ / BK)         // 16 K tiles
#define LDP 40                // padded LDS row pitch (bf16): 80B, 16B-aligned

// Low 32 bits of a generic pointer to LDS == LDS byte offset (flat aperture rule).
__device__ __forceinline__ unsigned lds_off(const void* p) {
    return (unsigned)(unsigned long long)p;
}

// Async 16B global->LDS copy, tracked by ASYNCcnt (no VGPR data path).
__device__ __forceinline__ void async_copy16(unsigned lds_addr, const __bf16* gaddr) {
    asm volatile("global_load_async_to_lds_b128 %0, %1, off"
                 :: "v"(lds_addr), "v"(gaddr) : "memory");
}
__device__ __forceinline__ void wait_async0() {
    asm volatile("s_wait_asynccnt 0x0" ::: "memory");
}

// Combine two 16-byte LDS loads into one v16bf fragment.
__device__ __forceinline__ v16bf ld_frag(const __bf16* lo, const __bf16* hi) {
    v8bf a = *(const v8bf*)lo;
    v8bf b = *(const v8bf*)hi;
    return __builtin_shufflevector(a, b, 0, 1, 2, 3, 4, 5, 6, 7,
                                         8, 9, 10, 11, 12, 13, 14, 15);
}

// ---- Phase 0a: X fp32 -> bf16 (layout preserved [M_, D_]) ------------------
__global__ __launch_bounds__(256)
void conv_x_kernel(const float* __restrict__ X, __bf16* __restrict__ Xbf) {
    const size_t i = (size_t)blockIdx.x * 256 + threadIdx.x;   // over float4s
    const float4 v = ((const float4*)X)[i];
    v4bf q;
    q[0] = (__bf16)v.x; q[1] = (__bf16)v.y;
    q[2] = (__bf16)v.z; q[3] = (__bf16)v.w;
    ((v4bf*)Xbf)[i] = q;
}

// ---- Phase 0b: W fp32 [d][k][n] -> bf16 transposed Wt [d][n][k] ------------
__global__ __launch_bounds__(256)
void conv_wt_kernel(const float* __restrict__ W, __bf16* __restrict__ Wt) {
    const size_t o = (size_t)blockIdx.x * 256 + threadIdx.x;   // k fastest
    const int k = (int)(o & (D_ - 1));
    const size_t nd = o >> 9;                                   // n + d*R_
    const int n = (int)(nd & (R_ - 1));
    const int d = (int)(nd >> 10);
    Wt[o] = (__bf16)W[((size_t)d * D_ + k) * R_ + n];
}

// ---- Phase 1: P[d] = Xbf @ Wt[d]^T  (bf16 WMMA) ----------------------------
__global__ __launch_bounds__(256)
void indrnn_gemm_kernel(const __bf16* __restrict__ Xbf,  // [M_, D_]
                        const __bf16* __restrict__ Wt,   // [DEPTH_, R_, D_]
                        float* __restrict__ P)           // [DEPTH_, M_, R_]
{
    __shared__ __bf16 As[2][BM * LDP];   // X tile [m][k], double buffered
    __shared__ __bf16 Bs[2][BN * LDP];   // Wt tile [n][k], double buffered

    const int tid  = threadIdx.x;
    const int lane = tid & 31;
    const int wave = tid >> 5;
    const int wm   = wave & 1;        // 2 wave rows * 64 M
    const int wn   = wave >> 1;       // 4 wave cols * 64 N
    const int half = lane >> 4;
    const int l15  = lane & 15;

    const int n0 = blockIdx.x * BN;
    const int m0 = blockIdx.y * BM;
    const int d  = blockIdx.z;

    const __bf16* Wd = Wt + (size_t)d * R_ * D_;
    float*        Pd = P  + (size_t)d * M_ * R_;

    // Staging slots (16B chunks, 4 per 32-wide K row):
    //   A tile: 128 rows * 4 = 512 chunks -> 2 per thread
    //   B tile: 256 rows * 4 = 1024 chunks -> 4 per thread
    const int arow0 = tid >> 2,           ac = tid & 3;
    const int arow1 = (tid + 256) >> 2;

    auto stage = [&](int k0, int buf) {
        async_copy16(lds_off(&As[buf][arow0 * LDP + ac * 8]),
                     Xbf + (size_t)(m0 + arow0) * D_ + k0 + ac * 8);
        async_copy16(lds_off(&As[buf][arow1 * LDP + ac * 8]),
                     Xbf + (size_t)(m0 + arow1) * D_ + k0 + ac * 8);
#pragma unroll
        for (int i = 0; i < 4; ++i) {
            const int brow = (tid + 256 * i) >> 2;
            async_copy16(lds_off(&Bs[buf][brow * LDP + ac * 8]),
                         Wd + (size_t)(n0 + brow) * D_ + k0 + ac * 8);
        }
    };

    const v8f vzero = {0.f, 0.f, 0.f, 0.f, 0.f, 0.f, 0.f, 0.f};
    v8f acc[4][4];                      // [ni][mi]
#pragma unroll
    for (int ni = 0; ni < 4; ++ni)
#pragma unroll
        for (int mi = 0; mi < 4; ++mi)
            acc[ni][mi] = vzero;

    stage(0, 0);
    wait_async0();
    __syncthreads();

    for (int kt = 0; kt < KT; ++kt) {
        const int buf = kt & 1;
        if (kt + 1 < KT) stage((kt + 1) * BK, buf ^ 1);   // prefetch next tile

        // Load 4 A fragments once; amortize each B fragment over 4 WMMAs.
        v16bf af[4];
#pragma unroll
        for (int mi = 0; mi < 4; ++mi) {
            // A layout: lane half 0 -> K 0-7 & 16-23 ; half 1 -> K 8-15 & 24-31
            const __bf16* base = &As[buf][(wm * 64 + mi * 16 + l15) * LDP + half * 8];
            af[mi] = ld_frag(base, base + 16);
        }
#pragma unroll
        for (int ni = 0; ni < 4; ++ni) {
            // B layout: lane half 0 -> K 0-7 & 16-23 ; half 1 -> K 8-15 & 24-31 of column l15
            const __bf16* base = &Bs[buf][(wn * 64 + ni * 16 + l15) * LDP + half * 8];
            const v16bf bfr = ld_frag(base, base + 16);
#pragma unroll
            for (int mi = 0; mi < 4; ++mi)   // 4 independent accumulators
                acc[ni][mi] = __builtin_amdgcn_wmma_f32_16x16x32_bf16(
                    false, af[mi], false, bfr,
                    (short)0, acc[ni][mi], false, false);
        }

        wait_async0();       // next tile landed (this wave's copies)
        __syncthreads();     // ... and every other wave's
    }

    // C layout: VGPR j holds M=j (lanes 0-15) / M=j+8 (lanes 16-31).
    // P streams once (256 MB > L2): nontemporal stores keep X/Wt resident.
    const int rb = m0 + wm * 64 + half * 8;
    const int cb = n0 + wn * 64 + l15;
#pragma unroll
    for (int ni = 0; ni < 4; ++ni)
#pragma unroll
        for (int mi = 0; mi < 4; ++mi)
#pragma unroll
            for (int j = 0; j < 8; ++j)
                __builtin_nontemporal_store(
                    acc[ni][mi][j],
                    &Pd[(size_t)(rb + mi * 16 + j) * R_ + (cb + ni * 16)]);
}

// ---- Phase 2: recurrent scan over T ----------------------------------------
__global__ __launch_bounds__(256)
void indrnn_scan_kernel(const float* __restrict__ P,     // [DEPTH_, M_, R_]
                        const float* __restrict__ h0,    // [B_, R_]
                        const float* __restrict__ u,     // [DEPTH_, R_]
                        const float* __restrict__ bias,  // [DEPTH_, R_]
                        float* __restrict__ seq,         // [B_, T_, R_]
                        float* __restrict__ hlast)       // [B_, R_]
{
    const int idx = blockIdx.x * blockDim.x + threadIdx.x;  // 0 .. B_*R_-1
    const int b = idx >> 10;
    const int r = idx & (R_ - 1);

    const float u0 = u[r],    u1 = u[R_ + r];
    const float c0 = bias[r], c1 = bias[R_ + r];
    const float* p0 = P + (size_t)b * T_ * R_ + r;
    const float* p1 = p0 + (size_t)M_ * R_;
    float*       o  = seq + (size_t)b * T_ * R_ + r;

    float h = h0[idx];
    for (int t = 0; t < T_; ++t) {
        const size_t off = (size_t)t * R_;
        const float a0 = __builtin_nontemporal_load(&p0[off]);
        const float a1 = __builtin_nontemporal_load(&p1[off]);
        h = tanhf(fmaf(h, u0, a0 + c0));
        h = tanhf(fmaf(h, u1, a1 + c1));
        __builtin_nontemporal_store(h, &o[off]);
    }
    hlast[idx] = h;
}

extern "C" void kernel_launch(void* const* d_in, const int* in_sizes, int n_in,
                              void* d_out, int out_size, void* d_ws, size_t ws_size,
                              hipStream_t stream) {
    (void)in_sizes; (void)n_in; (void)out_size; (void)ws_size;
    const float* x  = (const float*)d_in[0];   // [B, T, D]
    const float* h0 = (const float*)d_in[1];   // [B, R]
    const float* w  = (const float*)d_in[2];   // [DEPTH, D, R]
    const float* u  = (const float*)d_in[3];   // [DEPTH, R]
    const float* bb = (const float*)d_in[4];   // [DEPTH, R]

    float* seq   = (float*)d_out;                        // [B, T, R]
    float* hlast = seq + (size_t)B_ * T_ * R_;           // [B, R]

    // Workspace layout: P (256 MB) | Xbf (32 MB) | Wt (2 MB)
    float*  P   = (float*)d_ws;
    __bf16* Xbf = (__bf16*)((char*)d_ws + (size_t)DEPTH_ * M_ * R_ * sizeof(float));
    __bf16* Wtb = Xbf + (size_t)M_ * D_;

    conv_x_kernel<<<(M_ * D_ / 4) / 256, 256, 0, stream>>>(x, Xbf);
    conv_wt_kernel<<<(DEPTH_ * R_ * D_) / 256, 256, 0, stream>>>(w, Wtb);

    dim3 ggrid(R_ / BN, M_ / BM, DEPTH_);                // (4, 256, 2)
    indrnn_gemm_kernel<<<ggrid, 256, 0, stream>>>(Xbf, Wtb, P);

    indrnn_scan_kernel<<<(B_ * R_) / 256, 256, 0, stream>>>(P, h0, u, bb, seq, hlast);
}
